// AttentionBlock_6923487282555
// MI455X (gfx1250) — matrix operand order repaired and checked by reading, not compiled, not run on hardware
//
#include <hip/hip_runtime.h>
#include <hip/hip_bf16.h>
#include <math.h>

// ---------------- constants (match reference) ----------------
#define D_MODEL   1024
#define N_HEADS   4
#define HEAD_DIM  256
#define KV_DIM    512      // HEAD_DIM * N_KV_GROUPS
#define HPG       2        // heads per kv group
#define FFN       2730
#define FFN_NPAD  2736     // pad to multiple of 48 for NT=3 tiling
#define FFN_KPAD  2752     // pad to multiple of 32 for K loop
#define SEQ       4096
#define BATCH     2
#define M_ROWS    (BATCH * SEQ)   // 8192

typedef __attribute__((ext_vector_type(16))) __bf16 bf16x16;
typedef __attribute__((ext_vector_type(8)))  __bf16 bf16x8;
typedef __attribute__((ext_vector_type(8)))  float  f32x8;

__device__ inline bf16x16 frag_cat(uint4 lo, uint4 hi) {
    bf16x8 l = __builtin_bit_cast(bf16x8, lo);
    bf16x8 h = __builtin_bit_cast(bf16x8, hi);
    return __builtin_shufflevector(l, h, 0, 1, 2, 3, 4, 5, 6, 7,
                                         8, 9, 10, 11, 12, 13, 14, 15);
}

// =====================================================================
// fp32 -> bf16 conversion with zero padding (weights)
// =====================================================================
__global__ void cvt_pad_kernel(const float* __restrict__ in, __bf16* __restrict__ out,
                               int N, int K, int Npad, int Kpad) {
    size_t total = (size_t)Npad * (size_t)Kpad;
    for (size_t i = (size_t)blockIdx.x * blockDim.x + threadIdx.x; i < total;
         i += (size_t)gridDim.x * blockDim.x) {
        int n = (int)(i / (size_t)Kpad);
        int k = (int)(i % (size_t)Kpad);
        float v = (n < N && k < K) ? in[(size_t)n * K + k] : 0.0f;
        out[i] = (__bf16)v;
    }
}

__global__ void zfill_bf16_kernel(__bf16* __restrict__ p, int ld, int col0, int col1, int rows) {
    int ncols = col1 - col0;
    int i = blockIdx.x * blockDim.x + threadIdx.x;
    if (i < rows * ncols) {
        int r = i / ncols, c = col0 + i % ncols;
        p[(size_t)r * ld + c] = (__bf16)0.0f;
    }
}

// =====================================================================
// RMSNorm: x fp32 [rows, D_MODEL] -> out bf16
// =====================================================================
__global__ void rmsnorm_kernel(const float* __restrict__ x, const float* __restrict__ w,
                               __bf16* __restrict__ out) {
    const int D = D_MODEL;
    int row = blockIdx.x;
    const float* xr = x + (size_t)row * D;
    float ss = 0.0f;
    for (int i = threadIdx.x; i < D; i += blockDim.x) { float v = xr[i]; ss += v * v; }
#pragma unroll
    for (int off = 16; off >= 1; off >>= 1) ss += __shfl_xor(ss, off, 32);
    __shared__ float red[8];
    if ((threadIdx.x & 31) == 0) red[threadIdx.x >> 5] = ss;
    __syncthreads();
    float tot = red[0] + red[1] + red[2] + red[3] + red[4] + red[5] + red[6] + red[7];
    float inv = rsqrtf(tot / (float)D + 1e-6f);
    __bf16* orow = out + (size_t)row * D;
    for (int i = threadIdx.x; i < D; i += blockDim.x)
        orow[i] = (__bf16)(xr[i] * inv * w[i]);
}

// =====================================================================
// WMMA GEMM: C[M,Npad] = A[M,Kpad](bf16) * W[Npad,Kpad](bf16)^T
// Macro-tile per wave: MT=2 x NT tiles of 16x16. K-loop step 32.
// MODE 1: bf16 out
// MODE 2: f32 out + f32 residual (aux, ld = ldc)
// MODE 3: bf16 out = silu(gate(aux bf16, ldaux)) * acc
// MODE 4: bf16 TRANSPOSED out: C[(row/SEQ)*ldc + col][SEQ] <- value (for V^T)
// =====================================================================
template <int MODE, int NT>
__global__ void gemm_wmma_kernel(const __bf16* __restrict__ A, const __bf16* __restrict__ W,
                                 void* __restrict__ C, const void* __restrict__ aux,
                                 int M, int Kpad, int Npad, int ldc, int ldaux) {
    int gtid = blockIdx.x * blockDim.x + threadIdx.x;
    int wv   = gtid >> 5;
    int lane = threadIdx.x & 31;
    int ngrp = Npad / (16 * NT);
    int mgrp = M / 32;
    if (wv >= mgrp * ngrp) return;
    int tm = (wv / ngrp) * 2;
    int tn = (wv % ngrp) * NT;
    int ln = lane & 15, half = lane >> 4;

    const uint4* a4[2];
#pragma unroll
    for (int mi = 0; mi < 2; ++mi)
        a4[mi] = (const uint4*)(A + (size_t)((tm + mi) * 16 + ln) * Kpad);
    const uint4* w4[NT];
#pragma unroll
    for (int ni = 0; ni < NT; ++ni)
        w4[ni] = (const uint4*)(W + (size_t)((tn + ni) * 16 + ln) * Kpad);

    f32x8 acc[2][NT];
#pragma unroll
    for (int mi = 0; mi < 2; ++mi)
#pragma unroll
        for (int ni = 0; ni < NT; ++ni) acc[mi][ni] = (f32x8){};

    for (int k0 = 0; k0 < Kpad; k0 += 32) {
        int kq = k0 >> 3;                 // uint4 index of this 32-elem chunk
        bf16x16 av[2], bv[NT];
#pragma unroll
        for (int mi = 0; mi < 2; ++mi)    // A 16x32 layout: dwords {4h..}, {8+4h..}
            av[mi] = frag_cat(a4[mi][kq + half], a4[mi][kq + 2 + half]);
#pragma unroll
        for (int ni = 0; ni < NT; ++ni)   // B 32x16 layout: dwords {4h..}, {8+4h..}
            bv[ni] = frag_cat(w4[ni][kq + half], w4[ni][kq + 2 + half]);
#pragma unroll
        for (int mi = 0; mi < 2; ++mi)
#pragma unroll
            for (int ni = 0; ni < NT; ++ni)
                acc[mi][ni] = __builtin_amdgcn_wmma_f32_16x16x32_bf16(
                    false, av[mi], false, bv[ni], (short)0, acc[mi][ni], false, false);
    }

#pragma unroll
    for (int mi = 0; mi < 2; ++mi) {
#pragma unroll
        for (int ni = 0; ni < NT; ++ni) {
            int col = (tn + ni) * 16 + ln;
#pragma unroll
            for (int r = 0; r < 8; ++r) {
                int row = (tm + mi) * 16 + r + 8 * half;
                float v = acc[mi][ni][r];
                if (MODE == 1) {
                    ((__bf16*)C)[(size_t)row * ldc + col] = (__bf16)v;
                } else if (MODE == 2) {
                    v += ((const float*)aux)[(size_t)row * ldc + col];
                    ((float*)C)[(size_t)row * ldc + col] = v;
                } else if (MODE == 3) {
                    float g = (float)((const __bf16*)aux)[(size_t)row * ldaux + col];
                    float sg = g / (1.0f + __expf(-g));
                    ((__bf16*)C)[(size_t)row * ldc + col] = (__bf16)(sg * v);
                } else { // MODE 4: transposed store (V^T: [b][dim][seq])
                    int bb = row / SEQ, l = row % SEQ;
                    ((__bf16*)C)[((size_t)bb * ldc + col) * SEQ + l] = (__bf16)v;
                }
            }
        }
    }
}

// =====================================================================
// Flash attention (causal, GQA). One wave per (b, head, 16-query tile).
// 4 waves / 128-thread block. Q tile staged in LDS. V consumed transposed
// (Vt[b][dim][key]) so PV B-fragments are 2x b128 loads.
// __launch_bounds__(128, 1): allow full VGPR budget (oacc = 128 VGPRs must
// stay resident; default occupancy cap caused per-chunk scratch spills).
// =====================================================================
__global__ void __launch_bounds__(128, 1)
flash_attn_kernel(const __bf16* __restrict__ Q, const __bf16* __restrict__ K,
                  const __bf16* __restrict__ Vt, __bf16* __restrict__ O) {
    const float SCALE = 0.0625f;           // 256^-0.5
    const float NEGINF = -1.0e30f;
    const int L = SEQ;

    int gtid = blockIdx.x * blockDim.x + threadIdx.x;
    int wv   = gtid >> 5;
    int lane = threadIdx.x & 31;
    int wib  = threadIdx.x >> 5;           // 0..3

    int qtPerSeq = L / 16;                                  // 256
    int b    = wv / (N_HEADS * qtPerSeq);
    int rem  = wv % (N_HEADS * qtPerSeq);
    int hidx = rem / qtPerSeq;
    int qt   = rem % qtPerSeq;
    int g    = hidx / HPG;
    int q0   = qt * 16;

    int n = lane & 15, half = lane >> 4;

    __shared__ __align__(16) __bf16 qlds[4][16][256];   // 32 KB
    __shared__ float pbuf[4][16][33];                   // 8.4 KB

    // ---- stage Q tile (16x256 bf16) into LDS: lane -> row n, cols [half*128,+128)
    {
        const uint4* qsrc =
            (const uint4*)(Q + (size_t)(b * L + q0 + n) * D_MODEL + hidx * HEAD_DIM);
        uint4* qdst = (uint4*)&qlds[wib][n][half * 128];
#pragma unroll
        for (int i = 0; i < 16; ++i)
            qdst[i] = qsrc[half * 16 + i];
    }
    asm volatile("s_wait_dscnt 0" ::: "memory");

    f32x8 oacc[16];
#pragma unroll
    for (int t = 0; t < 16; ++t) oacc[t] = (f32x8){};
    float m_s[8], l_s[8];
#pragma unroll
    for (int r = 0; r < 8; ++r) { m_s[r] = NEGINF; l_s[r] = 0.0f; }

    const uint4* qrow4 = (const uint4*)&qlds[wib][n][0];   // A-layout row = lane&15

    // incremented base pointers (avoid per-chunk 64-bit remul)
    const __bf16* kptr  = K + (size_t)(b * L + n) * KV_DIM + g * HEAD_DIM;
    const __bf16* vtrow = Vt + (size_t)(b * KV_DIM + g * HEAD_DIM + n) * L;

    int nchunks = (q0 + 15) / 32 + 1;
    for (int kc = 0; kc < nchunks; ++kc) {
        int c0 = kc * 32;

        // ---- S = Q K^T for 32 keys (two 16x16 tiles), K-dim = 256 ----
        const uint4* k0p = (const uint4*)kptr;                 // rows c0+n
        const uint4* k1p = (const uint4*)(kptr + 16 * KV_DIM); // rows c0+16+n
        kptr += 32 * KV_DIM;
        f32x8 s0 = {}, s1 = {};
#pragma unroll
        for (int c = 0; c < 8; ++c) {
            bf16x16 qa = frag_cat(qrow4[c * 4 + half], qrow4[c * 4 + 2 + half]);
            bf16x16 b0 = frag_cat(k0p[c * 4 + half], k0p[c * 4 + 2 + half]);
            bf16x16 b1 = frag_cat(k1p[c * 4 + half], k1p[c * 4 + 2 + half]);
            s0 = __builtin_amdgcn_wmma_f32_16x16x32_bf16(false, qa, false, b0,
                                                         (short)0, s0, false, false);
            s1 = __builtin_amdgcn_wmma_f32_16x16x32_bf16(false, qa, false, b1,
                                                         (short)0, s1, false, false);
        }

        // ---- online softmax over this 32-key chunk ----
        float fac_r[8], p0[8], p1[8];
#pragma unroll
        for (int r = 0; r < 8; ++r) {
            int qi = q0 + r + 8 * half;
            float v0 = s0[r] * SCALE; if (c0 + n      > qi) v0 = NEGINF;
            float v1 = s1[r] * SCALE; if (c0 + 16 + n > qi) v1 = NEGINF;
            float mx = fmaxf(v0, v1);
#pragma unroll
            for (int off = 8; off >= 1; off >>= 1) mx = fmaxf(mx, __shfl_xor(mx, off, 16));
            float mnew = fmaxf(m_s[r], mx);
            float fac  = __expf(m_s[r] - mnew);
            m_s[r] = mnew;
            float e0 = __expf(v0 - mnew);
            float e1 = __expf(v1 - mnew);
            float rs = e0 + e1;
#pragma unroll
            for (int off = 8; off >= 1; off >>= 1) rs += __shfl_xor(rs, off, 16);
            l_s[r] = l_s[r] * fac + rs;
            fac_r[r] = fac; p0[r] = e0; p1[r] = e1;
        }
#pragma unroll
        for (int t = 0; t < 16; ++t)
#pragma unroll
            for (int r = 0; r < 8; ++r) oacc[t][r] *= fac_r[r];

        // ---- relayout P: C-layout -> A-layout via LDS ----
#pragma unroll
        for (int r = 0; r < 8; ++r) {
            pbuf[wib][r + 8 * half][n]      = p0[r];
            pbuf[wib][r + 8 * half][16 + n] = p1[r];
        }
        asm volatile("s_wait_dscnt 0" ::: "memory");
        bf16x16 pav;
#pragma unroll
        for (int j = 0; j < 8; ++j) {
            int ka = 2 * j + 16 * half;
            pav[2 * j]     = (__bf16)pbuf[wib][n][ka];
            pav[2 * j + 1] = (__bf16)pbuf[wib][n][ka + 1];
        }
        asm volatile("s_wait_dscnt 0" ::: "memory");

        // ---- O += P * V  (16 tiles across head_dim 256), Vt is key-contiguous ----
        const uint4* vt4 = (const uint4*)(vtrow + c0 + 16 * half);
#pragma unroll
        for (int t = 0; t < 16; ++t) {
            // dim row = g*HEAD_DIM + t*16 + n; offset t*16*L elems = t*(2*L) uint4
            bf16x16 bv = frag_cat(vt4[(size_t)t * 2 * L], vt4[(size_t)t * 2 * L + 1]);
            oacc[t] = __builtin_amdgcn_wmma_f32_16x16x32_bf16(false, pav, false, bv,
                                                              (short)0, oacc[t], false, false);
        }
    }

    // ---- epilogue: O / l, store bf16 ----
#pragma unroll
    for (int r = 0; r < 8; ++r) {
        float inv = 1.0f / l_s[r];
        size_t orow = (size_t)(b * L + q0 + r + 8 * half) * D_MODEL + hidx * HEAD_DIM;
#pragma unroll
        for (int t = 0; t < 16; ++t)
            O[orow + t * 16 + n] = (__bf16)(oacc[t][r] * inv);
    }
}

// =====================================================================
// host orchestration
// =====================================================================
extern "C" void kernel_launch(void* const* d_in, const int* in_sizes, int n_in,
                              void* d_out, int out_size, void* d_ws, size_t ws_size,
                              hipStream_t stream) {
    const float* x    = (const float*)d_in[0];
    const float* w_na = (const float*)d_in[1];
    const float* Wq   = (const float*)d_in[2];
    const float* Wk   = (const float*)d_in[3];
    const float* Wv   = (const float*)d_in[4];
    const float* Wo   = (const float*)d_in[5];
    const float* w_nf = (const float*)d_in[6];
    const float* Wg   = (const float*)d_in[7];
    const float* Wu   = (const float*)d_in[8];
    const float* Wd   = (const float*)d_in[9];
    float* out = (float*)d_out;

    char* ws = (char*)d_ws;
    size_t off = 0;
    auto alloc = [&](size_t bytes) { size_t o = off; off += (bytes + 255) & ~(size_t)255; return o; };

    size_t wqO = alloc((size_t)D_MODEL * D_MODEL * 2);
    size_t wkO = alloc((size_t)KV_DIM * D_MODEL * 2);
    size_t wvO = alloc((size_t)KV_DIM * D_MODEL * 2);
    size_t woO = alloc((size_t)D_MODEL * D_MODEL * 2);
    size_t wgO = alloc((size_t)FFN_NPAD * D_MODEL * 2);
    size_t wuO = alloc((size_t)FFN_NPAD * D_MODEL * 2);
    size_t wdO = alloc((size_t)D_MODEL * FFN_KPAD * 2);
    size_t xnO = alloc((size_t)M_ROWS * D_MODEL * 2);      // also attn output (aliased)
    size_t qO  = alloc((size_t)M_ROWS * D_MODEL * 2);      // also xn2 (aliased)
    size_t kO  = alloc((size_t)M_ROWS * KV_DIM * 2);
    size_t vO  = alloc((size_t)M_ROWS * KV_DIM * 2);       // holds V^T [B][KV_DIM][SEQ]
    size_t xaO = alloc((size_t)M_ROWS * D_MODEL * 4);
    size_t gtO = alloc((size_t)M_ROWS * FFN_NPAD * 2);
    size_t hO  = alloc((size_t)M_ROWS * FFN_KPAD * 2);
    (void)ws_size;

    __bf16* wq_b = (__bf16*)(ws + wqO);
    __bf16* wk_b = (__bf16*)(ws + wkO);
    __bf16* wv_b = (__bf16*)(ws + wvO);
    __bf16* wo_b = (__bf16*)(ws + woO);
    __bf16* wg_b = (__bf16*)(ws + wgO);
    __bf16* wu_b = (__bf16*)(ws + wuO);
    __bf16* wd_b = (__bf16*)(ws + wdO);
    __bf16* xn_b = (__bf16*)(ws + xnO);
    __bf16* attn_b = xn_b;
    __bf16* q_b  = (__bf16*)(ws + qO);
    __bf16* xn2_b = q_b;
    __bf16* k_b  = (__bf16*)(ws + kO);
    __bf16* vt_b = (__bf16*)(ws + vO);
    float*  xa_f = (float*)(ws + xaO);
    __bf16* gt_b = (__bf16*)(ws + gtO);
    __bf16* h_b  = (__bf16*)(ws + hO);

    const dim3 blk(256);
    const dim3 cvtGrid(2048);

    cvt_pad_kernel<<<cvtGrid, blk, 0, stream>>>(Wq, wq_b, D_MODEL, D_MODEL, D_MODEL, D_MODEL);
    cvt_pad_kernel<<<cvtGrid, blk, 0, stream>>>(Wk, wk_b, KV_DIM, D_MODEL, KV_DIM, D_MODEL);
    cvt_pad_kernel<<<cvtGrid, blk, 0, stream>>>(Wv, wv_b, KV_DIM, D_MODEL, KV_DIM, D_MODEL);
    cvt_pad_kernel<<<cvtGrid, blk, 0, stream>>>(Wo, wo_b, D_MODEL, D_MODEL, D_MODEL, D_MODEL);
    cvt_pad_kernel<<<cvtGrid, blk, 0, stream>>>(Wg, wg_b, FFN, D_MODEL, FFN_NPAD, D_MODEL);
    cvt_pad_kernel<<<cvtGrid, blk, 0, stream>>>(Wu, wu_b, FFN, D_MODEL, FFN_NPAD, D_MODEL);
    cvt_pad_kernel<<<cvtGrid, blk, 0, stream>>>(Wd, wd_b, D_MODEL, FFN, D_MODEL, FFN_KPAD);

    rmsnorm_kernel<<<dim3(M_ROWS), blk, 0, stream>>>(x, w_na, xn_b);

    auto gemmGrid = [](int M, int Npad, int NT) {
        int waves = (M / 32) * (Npad / (16 * NT));
        return dim3((unsigned)((waves + 7) / 8));
    };

    // Q/K projections (bf16 out), V projection stored transposed (MODE 4)
    gemm_wmma_kernel<1, 4><<<gemmGrid(M_ROWS, D_MODEL, 4), blk, 0, stream>>>(
        xn_b, wq_b, q_b, nullptr, M_ROWS, D_MODEL, D_MODEL, D_MODEL, 0);
    gemm_wmma_kernel<1, 4><<<gemmGrid(M_ROWS, KV_DIM, 4), blk, 0, stream>>>(
        xn_b, wk_b, k_b, nullptr, M_ROWS, D_MODEL, KV_DIM, KV_DIM, 0);
    gemm_wmma_kernel<4, 4><<<gemmGrid(M_ROWS, KV_DIM, 4), blk, 0, stream>>>(
        xn_b, wv_b, vt_b, nullptr, M_ROWS, D_MODEL, KV_DIM, KV_DIM, 0);

    // causal GQA flash attention (4 waves / 128-thread block)
    {
        int waves = BATCH * N_HEADS * (SEQ / 16);           // 2048
        flash_attn_kernel<<<dim3(waves / 4), dim3(128), 0, stream>>>(q_b, k_b, vt_b, attn_b);
    }

    // O projection + residual -> x_attn fp32
    gemm_wmma_kernel<2, 4><<<gemmGrid(M_ROWS, D_MODEL, 4), blk, 0, stream>>>(
        attn_b, wo_b, xa_f, x, M_ROWS, D_MODEL, D_MODEL, D_MODEL, 0);

    rmsnorm_kernel<<<dim3(M_ROWS), blk, 0, stream>>>(xa_f, w_nf, xn2_b);

    // gate = xn2 @ Wg^T (bf16, pre-silu)
    gemm_wmma_kernel<1, 3><<<gemmGrid(M_ROWS, FFN_NPAD, 3), blk, 0, stream>>>(
        xn2_b, wg_b, gt_b, nullptr, M_ROWS, D_MODEL, FFN_NPAD, FFN_NPAD, 0);

    // zero h tail cols [FFN_NPAD, FFN_KPAD)
    {
        int elems = M_ROWS * (FFN_KPAD - FFN_NPAD);
        zfill_bf16_kernel<<<dim3((elems + 255) / 256), blk, 0, stream>>>(
            h_b, FFN_KPAD, FFN_NPAD, FFN_KPAD, M_ROWS);
    }

    // up GEMM fused with h = silu(gate) * up
    gemm_wmma_kernel<3, 3><<<gemmGrid(M_ROWS, FFN_NPAD, 3), blk, 0, stream>>>(
        xn2_b, wu_b, h_b, gt_b, M_ROWS, D_MODEL, FFN_NPAD, FFN_KPAD, FFN_NPAD);

    // down projection + residual -> d_out fp32
    gemm_wmma_kernel<2, 4><<<gemmGrid(M_ROWS, D_MODEL, 4), blk, 0, stream>>>(
        h_b, wd_b, out, xa_f, M_ROWS, FFN_KPAD, D_MODEL, D_MODEL, 0);

    (void)in_sizes; (void)n_in; (void)out_size;
}
